// CoarseMatching_76381698392656
// MI455X (gfx1250) — matrix operand order repaired and checked by reading, not compiled, not run on hardware
//
#include <hip/hip_runtime.h>
#include <stdint.h>

// ---------------- CDNA5 (gfx1250) fused GEMM + exp + top-k + row-argmax ----
// sim = ref @ src^T via split-bf16 WMMA (hi/lo, 3x V_WMMA_F32_16X16X32_BF16).
// Software-pipelined: global->reg loads + f32->bf16(hi/lo) conversion for
// K-chunk k+1 overlap the WMMAs of chunk k (conversions co-execute with XDL).
// Workspace layout (assumes ws_size >= ~132 KB):
//   [0)              u64 rowmax[N]          (packed fbits<<32 | ~col)
//   [N*8)            u32 hist[4096]
//   [+16K)           u32 tilemax[4096]
//   [+16K)           u32 counters[2]        {cand_count, thr_bits}
//   [+16)            u64 cand[4096]

#define TILE       128
#define KC         32          // bf16 WMMA K
#define CDIM       256
#define NITER      (CDIM / KC) // 8 pipeline stages
#define LDS_STRIDE 40          // ushorts per tile row (32 data + 8 pad) = 80 B
#define NBUCKET    4096
#define CAND_CAP   4096
#define TOPK       256

typedef __attribute__((ext_vector_type(16))) __bf16 v16bf;
typedef __attribute__((ext_vector_type(8)))  float  v8f;

union Frag { v16bf v; uint4 q[2]; };

// Per-thread staging buffer: one 16-element K-slice of one A row and one B
// row, already split into packed bf16 hi/lo pairs (8 dwords per array).
struct Packed {
  uint4 ahi[2], alo[2], bhi[2], blo[2];
};

__device__ __forceinline__ unsigned bf16_rne(float x) {
  unsigned u = __float_as_uint(x);
  return (u + 0x7FFFu + ((u >> 16) & 1u)) >> 16;
}
__device__ __forceinline__ float bf16_tof(unsigned h) {
  return __uint_as_float(h << 16);
}
__device__ __forceinline__ unsigned pack_hi(float x0, float x1) {
  return bf16_rne(x0) | (bf16_rne(x1) << 16);
}
__device__ __forceinline__ unsigned pack_lo(float x0, float x1) {
  unsigned h0 = bf16_rne(x0), h1 = bf16_rne(x1);
  return bf16_rne(x0 - bf16_tof(h0)) | (bf16_rne(x1 - bf16_tof(h1)) << 16);
}

__device__ __forceinline__ void load_convert(const float* __restrict__ ga,
                                             const float* __restrict__ gb,
                                             Packed& p) {
  float xa[16], xb[16];
#pragma unroll
  for (int q = 0; q < 4; ++q) {
    float4 va = ((const float4*)ga)[q];
    float4 vb = ((const float4*)gb)[q];
    xa[4*q+0]=va.x; xa[4*q+1]=va.y; xa[4*q+2]=va.z; xa[4*q+3]=va.w;
    xb[4*q+0]=vb.x; xb[4*q+1]=vb.y; xb[4*q+2]=vb.z; xb[4*q+3]=vb.w;
  }
#pragma unroll
  for (int d = 0; d < 8; ++d) {     // dword d covers elements 2d, 2d+1
    unsigned* pa_hi = (unsigned*)&p.ahi[0];
    unsigned* pa_lo = (unsigned*)&p.alo[0];
    unsigned* pb_hi = (unsigned*)&p.bhi[0];
    unsigned* pb_lo = (unsigned*)&p.blo[0];
    pa_hi[d] = pack_hi(xa[2*d], xa[2*d+1]);
    pa_lo[d] = pack_lo(xa[2*d], xa[2*d+1]);
    pb_hi[d] = pack_hi(xb[2*d], xb[2*d+1]);
    pb_lo[d] = pack_lo(xb[2*d], xb[2*d+1]);
  }
}

// PASS 0: GEMM + exp + histogram + row-argmax + tile-max
// PASS 1: GEMM only for tiles above threshold; append candidates
template <int PASS>
__global__ __launch_bounds__(256)
void cm_gemm_kernel(const float* __restrict__ ref, const float* __restrict__ src,
                    int Mcols,
                    unsigned long long* __restrict__ rowmax,
                    unsigned* __restrict__ hist,
                    unsigned* __restrict__ tilemax,
                    unsigned* __restrict__ cand_count,
                    const unsigned* __restrict__ thrbits_p,
                    unsigned long long* __restrict__ cand)
{
  __shared__ __align__(16) unsigned short Ahi[TILE * LDS_STRIDE];
  __shared__ __align__(16) unsigned short Alo[TILE * LDS_STRIDE];
  __shared__ __align__(16) unsigned short Bhi[TILE * LDS_STRIDE];
  __shared__ __align__(16) unsigned short Blo[TILE * LDS_STRIDE];
  __shared__ unsigned lhist[NBUCKET];
  __shared__ unsigned long long lrow[TILE];
  __shared__ unsigned ltilemax;

  const int tid    = threadIdx.x;
  const int tile_n = blockIdx.x;          // src tile
  const int tile_m = blockIdx.y;          // ref tile
  const int tileId = tile_m * gridDim.x + tile_n;

  unsigned thr = 0;
  if (PASS == 1) {
    thr = thrbits_p[0];
    if (tilemax[tileId] < thr) return;    // uniform early exit for cold tiles
  }

  if (PASS == 0) {
    for (int b = tid; b < NBUCKET; b += 256) lhist[b] = 0;
    if (tid < TILE) lrow[tid] = 0ull;
    if (tid == 0) ltilemax = 0;
  }

  // ---- loader indexing: thread t stages row t>>1, 16 consecutive k ----
  const int lrow2 = tid >> 1;             // 0..127
  const int khalf = (tid & 1) * 16;       // 0 or 16
  const int lb    = lrow2 * LDS_STRIDE + khalf;  // byte ofs 80*r + {0,32}: 16B aligned
  // ---- compute indexing: 8 waves in 4x2 grid of 32x64 regions ----
  const int wave = tid >> 5;
  const int lane = tid & 31;
  const int wr   = wave >> 1;             // 0..3  (32-row group)
  const int wc   = wave & 1;              // 0..1  (64-col group)
  const int mloc = lane & 15;
  const int kbA  = (lane >> 4) << 3;      // A: lanes16-31 -> K base 8 (ISA 16b A 16x32)
  const int kbB  = (lane >> 4) << 3;      // B: lanes16-31 -> K base 8 (same K order as A)

  const v8f zacc = {0.f, 0.f, 0.f, 0.f, 0.f, 0.f, 0.f, 0.f};
  v8f acc[2][4];
#pragma unroll
  for (int i = 0; i < 2; ++i)
#pragma unroll
    for (int j = 0; j < 4; ++j) acc[i][j] = zacc;

  const float* gA = ref + (size_t)(tile_m * TILE + lrow2) * CDIM + khalf;
  const float* gB = src + (size_t)(tile_n * TILE + lrow2) * CDIM + khalf;

  Packed buf[2];
  load_convert(gA, gB, buf[0]);           // prologue: stage chunk 0

#pragma unroll
  for (int it = 0; it < NITER; ++it) {
    const int p = it & 1;
    __syncthreads();                      // previous compute done reading LDS
    // ------- registers -> LDS (2x ds_store_b128 per tile array) -------
    *(uint4*)&Ahi[lb]     = buf[p].ahi[0];
    *(uint4*)&Ahi[lb + 8] = buf[p].ahi[1];
    *(uint4*)&Alo[lb]     = buf[p].alo[0];
    *(uint4*)&Alo[lb + 8] = buf[p].alo[1];
    *(uint4*)&Bhi[lb]     = buf[p].bhi[0];
    *(uint4*)&Bhi[lb + 8] = buf[p].bhi[1];
    *(uint4*)&Blo[lb]     = buf[p].blo[0];
    *(uint4*)&Blo[lb + 8] = buf[p].blo[1];
    __syncthreads();                      // stores visible

    // ------- prefetch next chunk: overlaps the WMMAs below -------
    if (it + 1 < NITER)
      load_convert(gA + (it + 1) * KC, gB + (it + 1) * KC, buf[p ^ 1]);

    // ------- fragments (ds_load_b128 pairs, 16B aligned) -------
    Frag ahi[2], alo[2], bhi[4], blo[4];
#pragma unroll
    for (int i = 0; i < 2; ++i) {
      const int off = (wr * 32 + i * 16 + mloc) * LDS_STRIDE + kbA;
      ahi[i].q[0] = *(const uint4*)&Ahi[off];
      ahi[i].q[1] = *(const uint4*)&Ahi[off + 16];   // K += 16
      alo[i].q[0] = *(const uint4*)&Alo[off];
      alo[i].q[1] = *(const uint4*)&Alo[off + 16];
    }
#pragma unroll
    for (int j = 0; j < 4; ++j) {
      const int off = (wc * 64 + j * 16 + mloc) * LDS_STRIDE + kbB;
      bhi[j].q[0] = *(const uint4*)&Bhi[off];
      bhi[j].q[1] = *(const uint4*)&Bhi[off + 16];   // K += 16
      blo[j].q[0] = *(const uint4*)&Blo[off];
      blo[j].q[1] = *(const uint4*)&Blo[off + 16];
    }

    // ------- split-precision WMMA: hi*hi + hi*lo + lo*hi -------
#pragma unroll
    for (int i = 0; i < 2; ++i)
#pragma unroll
      for (int j = 0; j < 4; ++j) {
        acc[i][j] = __builtin_amdgcn_wmma_f32_16x16x32_bf16(
            false, ahi[i].v, false, bhi[j].v, (short)0, acc[i][j], false, false);
        acc[i][j] = __builtin_amdgcn_wmma_f32_16x16x32_bf16(
            false, ahi[i].v, false, blo[j].v, (short)0, acc[i][j], false, false);
        acc[i][j] = __builtin_amdgcn_wmma_f32_16x16x32_bf16(
            false, alo[i].v, false, bhi[j].v, (short)0, acc[i][j], false, false);
      }
  }

  // ---------------- fused epilogue ----------------
  const int hi8  = (lane >> 4) << 3;      // C/D: lanes16-31 -> rows +8
  const int col0 = tile_n * TILE + wc * 64;
  const int rlb  = wr * 32;               // local row base of this wave
  unsigned mymax = 0;

#pragma unroll
  for (int i = 0; i < 2; ++i)
#pragma unroll
    for (int j = 0; j < 4; ++j) {
      const int c = col0 + j * 16 + mloc;
#pragma unroll
      for (int v = 0; v < 8; ++v) {
        const int rl = rlb + i * 16 + v + hi8;            // 0..127
        const float s = __expf(2.0f * acc[i][j][v] - 2.0f);
        const unsigned fb = __float_as_uint(s);
        if (PASS == 0) {
          unsigned bkt = fb >> 20; bkt = bkt < NBUCKET ? bkt : NBUCKET - 1;
          atomicAdd(&lhist[bkt], 1u);
          const unsigned long long key =
              ((unsigned long long)fb << 32) | (unsigned)(~(unsigned)c);
          atomicMax(&lrow[rl], key);
          mymax = mymax > fb ? mymax : fb;
        } else {
          if (fb >= thr) {
            const unsigned pos = atomicAdd(cand_count, 1u);
            if (pos < CAND_CAP) {
              const unsigned flat =
                  (unsigned)(tile_m * TILE + rl) * (unsigned)Mcols + (unsigned)c;
              cand[pos] = ((unsigned long long)fb << 32) | (unsigned)(~flat);
            }
          }
        }
      }
    }

  if (PASS == 0) {
    atomicMax(&ltilemax, mymax);
    __syncthreads();
    for (int b = tid; b < NBUCKET; b += 256) {
      const unsigned cnt = lhist[b];
      if (cnt) atomicAdd(&hist[b], cnt);
    }
    if (tid < TILE) atomicMax(&rowmax[tile_m * TILE + tid], lrow[tid]);
    if (tid == 0) tilemax[tileId] = ltilemax;
  }
}

__global__ void cm_init_kernel(unsigned long long* rowmax, unsigned* hist,
                               unsigned* tilemax, unsigned* counters, int nrow) {
  const int i = blockIdx.x * blockDim.x + threadIdx.x;
  if (i < nrow) rowmax[i] = 0ull;
  if (i < NBUCKET) hist[i] = 0u;
  if (i < NBUCKET) tilemax[i] = 0u;
  if (i < 2) counters[i] = 0u;
}

__global__ void cm_scan_kernel(const unsigned* __restrict__ hist,
                               unsigned* __restrict__ thrbits, int k) {
  if (threadIdx.x == 0) {
    unsigned long long sum = 0;
    unsigned thr = 0;
    for (int b = NBUCKET - 1; b >= 0; --b) {
      sum += hist[b];
      if (sum >= (unsigned long long)k) { thr = (unsigned)b << 20; break; }
    }
    thrbits[0] = thr;
  }
}

__global__ __launch_bounds__(256)
void cm_sort_kernel(const unsigned long long* __restrict__ cand,
                    const unsigned* __restrict__ cand_count,
                    float* __restrict__ out, int Mcols, int k) {
  __shared__ unsigned long long keys[CAND_CAP];
  const int tid = threadIdx.x;
  unsigned n = cand_count[0];
  if (n > CAND_CAP) n = CAND_CAP;
  for (int i = tid; i < CAND_CAP; i += 256)
    keys[i] = (i < (int)n) ? cand[i] : 0ull;
  __syncthreads();
  // bitonic sort ascending; read from the tail for descending scores
  for (int kk = 2; kk <= CAND_CAP; kk <<= 1) {
    for (int j = kk >> 1; j > 0; j >>= 1) {
      for (int i = tid; i < CAND_CAP; i += 256) {
        const int ij = i ^ j;
        if (ij > i) {
          const bool up = ((i & kk) == 0);
          const unsigned long long a = keys[i], b = keys[ij];
          if (up ? (a > b) : (a < b)) { keys[i] = b; keys[ij] = a; }
        }
      }
      __syncthreads();
    }
  }
  int* oi = (int*)out;
  if (tid < k) {
    const unsigned long long key = keys[CAND_CAP - 1 - tid];
    const unsigned fb = (unsigned)(key >> 32);
    const unsigned flat = ~(unsigned)(key & 0xFFFFFFFFu);
    oi[tid]       = (int)(flat / (unsigned)Mcols);   // ref index
    oi[k + tid]   = (int)(flat % (unsigned)Mcols);   // src index
    out[2 * k + tid] = __uint_as_float(fb);          // score
  }
}

__global__ void cm_rows_kernel(const unsigned long long* __restrict__ rowmax,
                               int* __restrict__ out, int n) {
  const int i = blockIdx.x * blockDim.x + threadIdx.x;
  if (i < n) out[i] = (int)(~(unsigned)(rowmax[i] & 0xFFFFFFFFu));
}

extern "C" void kernel_launch(void* const* d_in, const int* in_sizes, int n_in,
                              void* d_out, int out_size, void* d_ws, size_t ws_size,
                              hipStream_t stream) {
  const float* ref = (const float*)d_in[0];
  const float* src = (const float*)d_in[1];
  const int C = CDIM;
  const int N = in_sizes[0] / C;   // 8192
  const int M = in_sizes[1] / C;   // 8192
  const int K = TOPK;              // num_proposal = 256

  char* ws = (char*)d_ws;
  unsigned long long* rowmax = (unsigned long long*)ws;
  size_t off = (size_t)N * 8;
  unsigned* hist = (unsigned*)(ws + off);            off += (size_t)NBUCKET * 4;
  unsigned* tilemax = (unsigned*)(ws + off);         off += (size_t)NBUCKET * 4;
  unsigned* counters = (unsigned*)(ws + off);        off += 16;
  unsigned long long* cand = (unsigned long long*)(ws + off);
  (void)ws_size; (void)n_in; (void)out_size;

  const dim3 grid(M / TILE, N / TILE);

  cm_init_kernel<<<(N + 255) / 256, 256, 0, stream>>>(rowmax, hist, tilemax,
                                                      counters, N);
  cm_gemm_kernel<0><<<grid, 256, 0, stream>>>(ref, src, M, rowmax, hist,
                                              tilemax, counters + 0,
                                              counters + 1, cand);
  cm_scan_kernel<<<1, 32, 0, stream>>>(hist, counters + 1, K);
  cm_gemm_kernel<1><<<grid, 256, 0, stream>>>(ref, src, M, rowmax, hist,
                                              tilemax, counters + 0,
                                              counters + 1, cand);
  cm_sort_kernel<<<1, 256, 0, stream>>>(cand, counters + 0, (float*)d_out, M, K);
  cm_rows_kernel<<<(N + 255) / 256, 256, 0, stream>>>(rowmax,
                                                      (int*)d_out + 3 * K, N);
}
